// VectorQuantizer_59768764891916
// MI455X (gfx1250) — matrix operand order repaired and checked by reading, not compiled, not run on hardware
//
#include <hip/hip_runtime.h>

typedef __attribute__((ext_vector_type(16))) _Float16 v16h;
typedef __attribute__((ext_vector_type(8)))  float    v8f;

// Async-DMA pointer types: pointee is a 16B int vector in AS1 (global) / AS3 (LDS).
typedef int v4i_vs __attribute__((vector_size(16)));
typedef __attribute__((address_space(1))) v4i_vs* g_v4i_p;
typedef __attribute__((address_space(3))) v4i_vs* l_v4i_p;

#define N_CODES 8192
#define C_DIM   256
#define N_ROWS  32768          // 8*4096
#define ROWS_PER_WG 128        // 8 waves * 16 rows
#define CODES_PER_TILE 32
#define N_TILES (N_CODES / CODES_PER_TILE)   // 256
#define LDS_STRIDE 264         // halves: 528 B pitch, 16B-aligned, conflict-free reads
#define EF_STRIDE  264         // padded f16 codebook pitch == LDS pitch (same IOFFSET both sides)
#define TILE_BYTES (CODES_PER_TILE * LDS_STRIDE * 2)   // 16896
#define CHUNK_BYTES (8 * LDS_STRIDE * 2)               // 4224 (8 rows per chunk)
#define ZQ_ELEMS (N_ROWS * C_DIM)

#if __has_builtin(__builtin_amdgcn_global_load_async_to_lds_b128)
#define HAVE_ASYNC_LDS 1
#else
#define HAVE_ASYNC_LDS 0
#endif

// ---------------------------------------------------------------------------
// Kernel 1: codebook -> f16 (padded pitch) and -0.5*||e||^2 bias.
// ---------------------------------------------------------------------------
__global__ __launch_bounds__(256) void vq_prep(const float* __restrict__ ew,
                                               _Float16* __restrict__ ef16,
                                               float* __restrict__ nhn) {
    const int code = blockIdx.x;
    const int t = threadIdx.x;
    float v = ew[code * C_DIM + t];
    ef16[code * EF_STRIDE + t] = (_Float16)v;
    __shared__ float red[256];
    red[t] = v * v;
    __syncthreads();
    for (int s = 128; s > 0; s >>= 1) {
        if (t < s) red[t] += red[t + s];
        __syncthreads();
    }
    if (t == 0) nhn[code] = -0.5f * red[0];
}

// ---------------------------------------------------------------------------
// Kernel 2: WMMA GEMM (scores = z.e^T - 0.5||e||^2) + running row argmax.
// Double-buffered LDS; codebook tiles streamed with GLOBAL_LOAD_ASYNC_TO_LDS
// (ASYNCcnt) when available, else a register pipeline. One barrier per tile.
// ---------------------------------------------------------------------------
__global__ __launch_bounds__(256, 1) void vq_argmax(const float* __restrict__ z,
                                                    const _Float16* __restrict__ ef16,
                                                    const float* __restrict__ nhn,
                                                    int* __restrict__ idxOut) {
    __shared__ __align__(16) _Float16 lbs[2 * CODES_PER_TILE * LDS_STRIDE];

    const int t     = threadIdx.x;
    const int wave  = t >> 5;
    const int lane  = t & 31;
    const int l16   = lane & 15;
    const int khalf = lane >> 4;
    const int rowBase = blockIdx.x * ROWS_PER_WG;
    const int rowM    = rowBase + wave * 16 + l16;

    // --- A fragments (16x256 f16 slab per wave), contiguous k order (k = 16*khalf + h), matching B.
    v16h a[8];
#pragma unroll
    for (int kk = 0; kk < 8; ++kk) {
#pragma unroll
        for (int h = 0; h < 16; ++h) {
            const int K = kk * 32 + khalf * 16 + h;
            a[kk][h] = (_Float16)z[rowM * C_DIM + K];
        }
    }

    float bestV[8];
    int   bestI[8];
#pragma unroll
    for (int r = 0; r < 8; ++r) { bestV[r] = -3.4e38f; bestI[r] = 0; }

    // Per-thread chunk: rows (t>>5)+i*8, 16B column (t&31); pitch matches LDS.
    const char* gb = (const char*)ef16 + ((t >> 5) * (EF_STRIDE * 2) + (t & 31) * 16);
    char* l0 = (char*)&lbs[(t >> 5) * LDS_STRIDE + (t & 31) * 8];
    char* l1 = l0 + TILE_BYTES;
    const _Float16* bb0 = &lbs[khalf * 16 + l16 * LDS_STRIDE];
    const _Float16* bb1 = bb0 + CODES_PER_TILE * LDS_STRIDE;
    const float* np = nhn + l16;

#if HAVE_ASYNC_LDS
    auto asyncTile = [&](char* dst) {
        g_v4i_p g = (g_v4i_p)(void*)gb;
        l_v4i_p l = (l_v4i_p)(void*)dst;
        __builtin_amdgcn_global_load_async_to_lds_b128(g, l, 0 * CHUNK_BYTES, 0);
        __builtin_amdgcn_global_load_async_to_lds_b128(g, l, 1 * CHUNK_BYTES, 0);
        __builtin_amdgcn_global_load_async_to_lds_b128(g, l, 2 * CHUNK_BYTES, 0);
        __builtin_amdgcn_global_load_async_to_lds_b128(g, l, 3 * CHUNK_BYTES, 0);
        gb += TILE_BYTES;
    };
#else
    uint4 stg[4];
    auto ldTile = [&]() {
#pragma unroll
        for (int i = 0; i < 4; ++i) stg[i] = *(const uint4*)(gb + i * CHUNK_BYTES);
        gb += TILE_BYTES;
    };
    auto stTile = [&](char* dst) {
#pragma unroll
        for (int i = 0; i < 4; ++i) *(uint4*)(dst + i * CHUNK_BYTES) = stg[i];
    };
#endif

    auto compute = [&](int cb, const _Float16* bb) {
        const float nb0 = np[cb];
        const float nb1 = np[cb + 16];
        v8f c0 = (v8f){0.f,0.f,0.f,0.f,0.f,0.f,0.f,0.f};
        v8f c1 = (v8f){0.f,0.f,0.f,0.f,0.f,0.f,0.f,0.f};
#pragma unroll
        for (int kk = 0; kk < 8; ++kk) {
            const v16h b0 = *(const v16h*)&bb[kk * 32];
            const v16h b1 = *(const v16h*)&bb[kk * 32 + 16 * LDS_STRIDE];
            c0 = __builtin_amdgcn_wmma_f32_16x16x32_f16(false, a[kk], false, b0, (short)0, c0, false, false);
            c1 = __builtin_amdgcn_wmma_f32_16x16x32_f16(false, a[kk], false, b1, (short)0, c1, false, false);
        }
#pragma unroll
        for (int r = 0; r < 8; ++r) {
            const float s0 = c0[r] + nb0;
            if (s0 > bestV[r]) { bestV[r] = s0; bestI[r] = cb + l16; }
            const float s1 = c1[r] + nb1;
            if (s1 > bestV[r]) { bestV[r] = s1; bestI[r] = cb + 16 + l16; }
        }
    };

#if HAVE_ASYNC_LDS
    // Prologue: tile 0 -> buf0 via async DMA.
    asyncTile(l0);
    asm volatile("s_wait_asynccnt 0x0" ::: "memory");
    __syncthreads();

    for (int ti = 0; ti < N_TILES; ti += 2) {
        // even step: read buf0 (tile ti), DMA tile ti+1 into idle buf1
        asyncTile(l1);                           // ti+1 < N_TILES always (ti <= 254)
        compute(ti * CODES_PER_TILE, bb0);
        asm volatile("s_wait_asynccnt 0x0" ::: "memory");
        __syncthreads();
        // odd step: read buf1 (tile ti+1), DMA tile ti+2 into idle buf0
        if (ti + 2 < N_TILES) asyncTile(l0);
        compute((ti + 1) * CODES_PER_TILE, bb1);
        asm volatile("s_wait_asynccnt 0x0" ::: "memory");
        __syncthreads();
    }
#else
    // Prologue: tile 0 -> buf0; stg holds tile 1.
    ldTile();
    stTile(l0);
    ldTile();
    __syncthreads();

    for (int ti = 0; ti < N_TILES; ti += 2) {
        stTile(l1);                               // tile ti+1 -> buf1 (idle)
        if (ti + 2 < N_TILES) ldTile();           // loads for tile ti+2
        compute(ti * CODES_PER_TILE, bb0);
        __syncthreads();
        if (ti + 2 < N_TILES) stTile(l0);         // tile ti+2 -> buf0 (idle)
        if (ti + 3 < N_TILES) ldTile();           // loads for tile ti+3
        compute((ti + 1) * CODES_PER_TILE, bb1);
        __syncthreads();
    }
#endif

    // --- Reduce across the 16 lanes holding each accumulator row.
#pragma unroll
    for (int r = 0; r < 8; ++r) {
        float v = bestV[r];
        int   i = bestI[r];
#pragma unroll
        for (int m = 8; m >= 1; m >>= 1) {
            const float v2 = __shfl_xor(v, m, 32);
            const int   i2 = __shfl_xor(i, m, 32);
            if (v2 > v || (v2 == v && i2 < i)) { v = v2; i = i2; }
        }
        bestV[r] = v; bestI[r] = i;
    }
    if (l16 == 0) {
        const int base = rowBase + wave * 16 + khalf * 8;
#pragma unroll
        for (int r = 0; r < 8; ++r) idxOut[base + r] = bestI[r];
    }
}

// ---------------------------------------------------------------------------
// Kernel 3: gather z_q (== z_q_st forward value), write idx as float,
// per-block partial sums of (z_q - z)^2.
// ---------------------------------------------------------------------------
__global__ __launch_bounds__(256) void vq_output(const float* __restrict__ z,
                                                 const float* __restrict__ ew,
                                                 const int* __restrict__ idxIn,
                                                 float* __restrict__ zq_out,
                                                 float* __restrict__ idx_out_f,
                                                 float* __restrict__ partial) {
    const int gid = blockIdx.x * 256 + threadIdx.x;
    const int row = (gid * 4) >> 8;
    const int col = (gid * 4) & 255;
    const int code = idxIn[row];
    const float4 e  = *(const float4*)&ew[code * C_DIM + col];
    const float4 zz = *(const float4*)&z[row * C_DIM + col];
    *(float4*)&zq_out[row * C_DIM + col] = e;
    const float dx = e.x - zz.x, dy = e.y - zz.y, dz = e.z - zz.z, dw = e.w - zz.w;
    float s = dx * dx + dy * dy + dz * dz + dw * dw;

    __shared__ float red[256];
    red[threadIdx.x] = s;
    __syncthreads();
    for (int st = 128; st > 0; st >>= 1) {
        if (threadIdx.x < st) red[threadIdx.x] += red[threadIdx.x + st];
        __syncthreads();
    }
    if (threadIdx.x == 0) partial[blockIdx.x] = red[0];
    if (col == 0) idx_out_f[row] = (float)code;
}

// ---------------------------------------------------------------------------
// Kernel 4: deterministic final reduction -> loss = 1.25 * mean((zq - z)^2).
// ---------------------------------------------------------------------------
__global__ __launch_bounds__(256) void vq_final(const float* __restrict__ partial,
                                                float* __restrict__ lossOut) {
    const int t = threadIdx.x;
    float s = 0.f;
#pragma unroll
    for (int i = 0; i < 32; ++i) s += partial[t + i * 256];
    __shared__ float red[256];
    red[t] = s;
    __syncthreads();
    for (int st = 128; st > 0; st >>= 1) {
        if (t < st) red[t] += red[t + st];
        __syncthreads();
    }
    if (t == 0) lossOut[0] = 1.25f * red[0] / (float)ZQ_ELEMS;
}

// ---------------------------------------------------------------------------
extern "C" void kernel_launch(void* const* d_in, const int* in_sizes, int n_in,
                              void* d_out, int out_size, void* d_ws, size_t ws_size,
                              hipStream_t stream) {
    const float* z  = (const float*)d_in[0];   // [8,4096,256] f32
    const float* ew = (const float*)d_in[1];   // [8192,256] f32
    float* out = (float*)d_out;

    // ws layout (padded f16 codebook pitch = EF_STRIDE halves)
    const size_t EF_BYTES = (size_t)N_CODES * EF_STRIDE * 2;   // 4,325,376 (256-aligned)
    char* ws = (char*)d_ws;
    _Float16* ef16    = (_Float16*)ws;
    float*    nhn     = (float*)(ws + EF_BYTES);
    int*      idxbuf  = (int*)  (ws + EF_BYTES + 32768);
    float*    partial = (float*)(ws + EF_BYTES + 32768 + 131072);

    float* zq    = out;                 // [32768*256]
    float* lossp = out + ZQ_ELEMS;      // [1]
    float* idxf  = out + ZQ_ELEMS + 1;  // [32768]

    vq_prep  <<<N_CODES, 256, 0, stream>>>(ew, ef16, nhn);
    vq_argmax<<<N_ROWS / ROWS_PER_WG, 256, 0, stream>>>(z, ef16, nhn, idxbuf);
    vq_output<<<ZQ_ELEMS / (4 * 256), 256, 0, stream>>>(z, ew, idxbuf, zq, idxf, partial);
    vq_final <<<1, 256, 0, stream>>>(partial, lossp);
}
